// GCNConvOnly_34196529610952
// MI455X (gfx1250) — matrix operand order repaired and checked by reading, not compiled, not run on hardware
//
#include <hip/hip_runtime.h>
#include <hip/hip_bf16.h>
#include <math.h>

// ---------------------------------------------------------------------------
// GCN (3x GCNConv + ELU) for MI455X / gfx1250, wave32.
//  - Layer-1 dense transform uses v_wmma_f32_16x16x32_f16 (f32 accumulate),
//    K loop fully unrolled (8 WMMAs), branch-free A/B fragment construction.
//  - Edge aggregation uses global fp32 atomics (L2-resident feature arrays).
// ---------------------------------------------------------------------------

typedef __attribute__((ext_vector_type(16))) _Float16 v16h;
typedef __attribute__((ext_vector_type(8)))  float    v8f;

#define FIN 256
#define H1  8
#define H2  4
#define CC  16

__device__ __forceinline__ float elu1(float v) {
    return v > 0.0f ? v : expm1f(v);
}

// ---- degree / normalization -------------------------------------------------
__global__ void deg_init_kernel(float* __restrict__ deg, int N) {
    int i = blockIdx.x * blockDim.x + threadIdx.x;
    if (i < N) deg[i] = 1.0f;  // self loop weight 1
}

__global__ void deg_accum_kernel(const int* __restrict__ dst,
                                 const float* __restrict__ w,
                                 float* __restrict__ deg, int E) {
    int e = blockIdx.x * blockDim.x + threadIdx.x;
    if (e < E) atomicAdd(&deg[dst[e]], w[e]);
}

__global__ void dis_finalize_kernel(float* __restrict__ deg, int N) {
    int i = blockIdx.x * blockDim.x + threadIdx.x;
    if (i < N) deg[i] = rsqrtf(deg[i]);   // deg >= 1 always (self loop)
}

// ---- layer-1 GEMM: h1[N,8] = x[N,256] @ W1[256,8] via WMMA ------------------
// One wave computes a 16(row) x 16(col; cols 8..15 zero-padded) tile.
// Block = 256 threads = 8 waves = 128 rows. B fragments are pre-packed in LDS
// in the exact per-lane v16h layout so the hot loop is branch-free.
__global__ void gemm1_wmma_kernel(const float* __restrict__ x,
                                  const float* __restrict__ W1,
                                  float* __restrict__ h1, int N) {
    // 8 K-blocks x 32 lanes x 16 halfwords = 8 KB, 32B-aligned per lane slice
    __shared__ __align__(32) _Float16 sB[8 * 32 * 16];
    {
        // Thread t packs the B fragment for K-block (t>>5), lane (t&31).
        // Branch-free: clamp column index, mask padding columns to zero.
        const int t    = threadIdx.x;
        const int kb   = t >> 5;
        const int l    = t & 31;
        const int hf   = l >> 4;       // 0: K {0..7,16..23} of block, 1: K {8..15,24..31}
        const int n    = l & 15;       // output column this lane carries
        const int nc   = (n < H1) ? n : (H1 - 1);
        const float mk = (n < H1) ? 1.0f : 0.0f;
        _Float16* dp = &sB[t * 16];
        #pragma unroll
        for (int i = 0; i < 16; ++i) {
            int k = kb * 32 + hf * 8 + i + ((i >= 8) ? 8 : 0);
            dp[i] = (_Float16)(W1[k * H1 + nc] * mk);
        }
    }
    __syncthreads();

    const int lane = threadIdx.x & 31;
    const int wave = threadIdx.x >> 5;
    const int rowBase = blockIdx.x * 128 + wave * 16;
    const int half = lane >> 4;        // 0: lanes 0-15, 1: lanes 16-31
    const int mrow = lane & 15;        // A row-in-tile AND C/D column
    const int arow = rowBase + mrow;   // global A row for this lane
    // Branch-free OOB handling: clamp the address; results of clamped rows
    // are never stored, so loading row N-1's data is harmless.
    const int arow_c = (arow < N) ? arow : (N - 1);
    const float* xrow = x + (size_t)arow_c * FIN + half * 8;

    v8f c;
    #pragma unroll
    for (int j = 0; j < 8; ++j) c[j] = 0.0f;

    #pragma unroll
    for (int kb = 0; kb < 8; ++kb) {
        // A (16-bit 16x32 layout): lanes 0-15 hold K=kk+{0..7,16..23},
        // lanes 16-31 hold K=kk+{8..15,24..31}; two 32B contiguous groups.
        const float* xr = xrow + kb * 32;
        v16h a;
        #pragma unroll
        for (int i = 0; i < 8; ++i) a[i]     = (_Float16)xr[i];
        #pragma unroll
        for (int i = 0; i < 8; ++i) a[8 + i] = (_Float16)xr[16 + i];
        // B fragment: one aligned 32-byte LDS read, no branches.
        v16h b = *(const v16h*)&sB[(kb * 32 + lane) * 16];
        c = __builtin_amdgcn_wmma_f32_16x16x32_f16(
                /*neg_a=*/false, a, /*neg_b=*/false, b,
                /*c_mod=*/(short)0, c, /*reuse_a=*/false, /*reuse_b=*/false);
    }

    // D layout: VGPR j -> M = (half ? j+8 : j), N = mrow
    if (mrow < H1) {
        if (rowBase + 16 <= N) {   // fast path: whole tile in range
            #pragma unroll
            for (int j = 0; j < 8; ++j) {
                int r = rowBase + (half ? j + 8 : j);
                h1[(size_t)r * H1 + mrow] = c[j];
            }
        } else {
            #pragma unroll
            for (int j = 0; j < 8; ++j) {
                int r = rowBase + (half ? j + 8 : j);
                if (r < N) h1[(size_t)r * H1 + mrow] = c[j];
            }
        }
    }
}

// ---- tiny VALU GEMMs (layers 2/3), ELU fused on the input -------------------
template <int KIN, int NOUT>
__global__ void gemm_small_kernel(const float* __restrict__ a,
                                  const float* __restrict__ W,
                                  float* __restrict__ h, int N) {
    int i = blockIdx.x * blockDim.x + threadIdx.x;
    if (i >= N) return;
    float av[KIN];
    #pragma unroll
    for (int k = 0; k < KIN; ++k) av[k] = elu1(a[(size_t)i * KIN + k]);
    #pragma unroll
    for (int n = 0; n < NOUT; ++n) {
        float acc = 0.0f;
        #pragma unroll
        for (int k = 0; k < KIN; ++k) acc = fmaf(av[k], W[k * NOUT + n], acc);
        h[(size_t)i * NOUT + n] = acc;
    }
}

// ---- aggregation: init = bias + self-loop contribution ----------------------
template <int FW>
__global__ void agg_init_kernel(const float* __restrict__ m,
                                const float* __restrict__ dis,
                                const float* __restrict__ bias,
                                float* __restrict__ out, int N) {
    int i = blockIdx.x * blockDim.x + threadIdx.x;
    if (i >= N) return;
    float sl = dis[i] * dis[i];  // dis[i] * 1 * dis[i]
    #pragma unroll
    for (int f = 0; f < FW; ++f)
        out[(size_t)i * FW + f] = bias[f] + sl * m[(size_t)i * FW + f];
}

// ---- aggregation: per-edge gather/scale/scatter-add -------------------------
template <int FW>
__global__ void agg_edges_kernel(const int* __restrict__ src,
                                 const int* __restrict__ dst,
                                 const float* __restrict__ w,
                                 const float* __restrict__ dis,
                                 const float* __restrict__ m,
                                 float* __restrict__ out, int E) {
    int e = blockIdx.x * blockDim.x + threadIdx.x;
    if (e >= E) return;
    int s = src[e], d = dst[e];
    float nrm = dis[s] * w[e] * dis[d];
    const float* ms = m + (size_t)s * FW;
    float* od = out + (size_t)d * FW;
    #pragma unroll
    for (int f = 0; f < FW; ++f) atomicAdd(&od[f], nrm * ms[f]);
}

// ---------------------------------------------------------------------------
extern "C" void kernel_launch(void* const* d_in, const int* in_sizes, int n_in,
                              void* d_out, int out_size, void* d_ws, size_t ws_size,
                              hipStream_t stream) {
    const float* x  = (const float*)d_in[0];
    const int*   ei = (const int*)d_in[1];   // edge_index [2,E] (int32 on device)
    const float* ew = (const float*)d_in[2];
    const float* W1 = (const float*)d_in[3];
    const float* b1 = (const float*)d_in[4];
    const float* W2 = (const float*)d_in[5];
    const float* b2 = (const float*)d_in[6];
    const float* W3 = (const float*)d_in[7];
    const float* b3 = (const float*)d_in[8];

    const int N = in_sizes[0] / FIN;
    const int E = in_sizes[2];
    const int* src = ei;
    const int* dst = ei + E;

    // workspace layout (floats): dis[N] | h1[8N] | a1[8N] | h2[4N] | a2[4N] | h3[16N]
    float* ws  = (float*)d_ws;
    float* dis = ws;
    float* h1  = dis + (size_t)N;
    float* a1  = h1  + (size_t)N * H1;
    float* h2  = a1  + (size_t)N * H1;
    float* a2  = h2  + (size_t)N * H2;
    float* h3  = a2  + (size_t)N * H2;
    float* out = (float*)d_out;

    const dim3 blk(256);
    const int gN = (N + 255) / 256;
    const int gE = (E + 255) / 256;
    const int gG = (N + 127) / 128;  // gemm1: 128 rows per block

    // normalization
    deg_init_kernel<<<gN, blk, 0, stream>>>(dis, N);
    deg_accum_kernel<<<gE, blk, 0, stream>>>(dst, ew, dis, E);
    dis_finalize_kernel<<<gN, blk, 0, stream>>>(dis, N);

    // layer 1: WMMA dense transform, then aggregate
    gemm1_wmma_kernel<<<gG, blk, 0, stream>>>(x, W1, h1, N);
    agg_init_kernel<H1><<<gN, blk, 0, stream>>>(h1, dis, b1, a1, N);
    agg_edges_kernel<H1><<<gE, blk, 0, stream>>>(src, dst, ew, dis, h1, a1, E);

    // layer 2: ELU fused into tiny GEMM, then aggregate
    gemm_small_kernel<H1, H2><<<gN, blk, 0, stream>>>(a1, W2, h2, N);
    agg_init_kernel<H2><<<gN, blk, 0, stream>>>(h2, dis, b2, a2, N);
    agg_edges_kernel<H2><<<gE, blk, 0, stream>>>(src, dst, ew, dis, h2, a2, E);

    // layer 3: ELU fused, final aggregate straight into d_out (no ELU on output)
    gemm_small_kernel<H2, CC><<<gN, blk, 0, stream>>>(a2, W3, h3, N);
    agg_init_kernel<CC><<<gN, blk, 0, stream>>>(h3, dis, b3, out, N);
    agg_edges_kernel<CC><<<gE, blk, 0, stream>>>(src, dst, ew, dis, h3, out, E);
}
